// MilliesRNN_11244224381288
// MI455X (gfx1250) — matrix operand order repaired and checked by reading, not compiled, not run on hardware
//
#include <hip/hip_runtime.h>

#define B_   64
#define T_   512
#define I_   256
#define H_   1024
#define O_   256
#define NWG  64
#define NTHR 128

typedef __attribute__((ext_vector_type(16))) _Float16 v16h;
typedef __attribute__((ext_vector_type(8)))  _Float16 v8h;
typedef __attribute__((ext_vector_type(8)))  float    v8f;

// ---- WMMA operand loaders (contiguous k per lane half, k = 16h + s, wave32) ----

// A operand, 16x32 f16 tile from row-major f16 matrix (ld = row stride in elems).
// lane l: row m0+(l&15); K = kb..kb+15 contiguous, kb = k0 + (l>=16 ? 16 : 0)
static __device__ __forceinline__ v16h load_a_f16(const _Float16* __restrict__ base,
                                                  int ld, int m0, int k0, int lane) {
  int r  = m0 + (lane & 15);
  int kb = k0 + ((lane & 16) ? 16 : 0);
  const _Float16* p = base + r * ld + kb;
  v8h lo = *(const v8h*)(p);
  v8h hi = *(const v8h*)(p + 8);
  v16h a;
#pragma unroll
  for (int i = 0; i < 8; ++i) { a[i] = lo[i]; a[i + 8] = hi[i]; }
  return a;
}

// Same A layout but source is f32 (x_t), converted in-register to f16.
static __device__ __forceinline__ v16h load_a_f32(const float* __restrict__ base,
                                                  int ld, int m0, int k0, int lane) {
  int r  = m0 + (lane & 15);
  int kb = k0 + ((lane & 16) ? 16 : 0);
  const float* p = base + (long)r * ld + kb;
  v8f lo = *(const v8f*)(p);
  v8f hi = *(const v8f*)(p + 8);
  v16h a;
#pragma unroll
  for (int i = 0; i < 8; ++i) { a[i] = (_Float16)lo[i]; a[i + 8] = (_Float16)hi[i]; }
  return a;
}

// B operand, 32x16 f16 tile. We compute act @ W^T with row-major W[N,K] (ld = K):
// B[k][n] = W[n][k] -> lane l: col n0+(l&15), 16 contiguous halves at
// k0 + (l>=16 ? 16 : 0). One 32-byte load per lane.
static __device__ __forceinline__ v16h load_b_f16(const _Float16* __restrict__ W,
                                                  int ld, int n0, int k0, int lane) {
  int n  = n0 + (lane & 15);
  int kb = k0 + ((lane & 16) ? 16 : 0);
  return *(const v16h*)(W + n * ld + kb);
}

static __device__ __forceinline__ v8f wmma_f16(v8f acc, v16h a, v16h b) {
  return __builtin_amdgcn_wmma_f32_16x16x32_f16(false, a, false, b,
                                                (short)0, acc, false, false);
}

static __device__ __forceinline__ float retanh(float x) {
  return tanhf(fmaxf(x, 0.0f));
}

// Device-scope sense-free barrier: monotone counter, per-instance target derived
// from arrival value. Counter reset each launch via hipMemsetAsync.
static __device__ __forceinline__ void gsync(unsigned* bar) {
  __syncthreads();
  if (threadIdx.x == 0) {
    __threadfence();
    unsigned arrive = __hip_atomic_fetch_add(bar, 1u, __ATOMIC_ACQ_REL,
                                             __HIP_MEMORY_SCOPE_AGENT);
    unsigned target = (arrive / NWG + 1u) * NWG;
    while (__hip_atomic_load(bar, __ATOMIC_ACQUIRE, __HIP_MEMORY_SCOPE_AGENT) < target)
      __builtin_amdgcn_s_sleep(1);
  }
  __syncthreads();
}

// Split-K reduction across the 4 waves through LDS; wave 0 finishes the tile.
static __device__ __forceinline__ void reduce4_store_f16(
    v8f acc, float red[4][32][8], int wave, int lane,
    _Float16* __restrict__ dst, int ldd, int m0, int n0,
    const float* __restrict__ bias, bool act) {
#pragma unroll
  for (int v = 0; v < 8; ++v) red[wave][lane][v] = acc[v];
  __syncthreads();
  if (wave == 0) {
    int n  = n0 + (lane & 15);
    int mb = m0 + ((lane & 16) ? 8 : 0);
    float bb = bias[n];
#pragma unroll
    for (int v = 0; v < 8; ++v) {
      float x = red[0][lane][v] + red[1][lane][v] + red[2][lane][v] + red[3][lane][v] + bb;
      if (act) x = retanh(x);
      dst[(mb + v) * ldd + n] = (_Float16)x;
    }
  }
}

static __device__ __forceinline__ void reduce4_store_out(
    v8f acc, float red[4][32][8], int wave, int lane,
    float* __restrict__ out, int t, int m0, int n0,
    const float* __restrict__ bias) {
#pragma unroll
  for (int v = 0; v < 8; ++v) red[wave][lane][v] = acc[v];
  __syncthreads();
  if (wave == 0) {
    int n  = n0 + (lane & 15);
    int mb = m0 + ((lane & 16) ? 8 : 0);
    float bb = bias[n];
#pragma unroll
    for (int v = 0; v < 8; ++v) {
      float x = red[0][lane][v] + red[1][lane][v] + red[2][lane][v] + red[3][lane][v] + bb;
      out[(mb + v) * (T_ * O_) + t * O_ + n] = x;
    }
  }
}

__global__ __launch_bounds__(NTHR) void millies_rnn_persistent(
    const float* __restrict__ data,
    const float* __restrict__ bi2h,     const float* __restrict__ bh2h,
    const float* __restrict__ bh2o,     const float* __restrict__ bthal,
    const float* __restrict__ bi2h_dos, const float* __restrict__ bh2h_dos,
    const float* __restrict__ bh2o_dos,
    const _Float16* __restrict__ Wi2h,     const _Float16* __restrict__ Wh2h,
    const _Float16* __restrict__ Wh2o,     const _Float16* __restrict__ Wthal,
    const _Float16* __restrict__ Wi2h_dos, const _Float16* __restrict__ Wh2h_dos,
    const _Float16* __restrict__ Wh2o_dos,
    _Float16* hv0, _Float16* hv1, _Float16* hm0, _Float16* hm1,
    _Float16* out_v, _Float16* out_t,
    float* __restrict__ out, unsigned* bar)
{
  __shared__ float red[4][32][8];
  const int g    = blockIdx.x;         // 0..63
  const int wave = threadIdx.x >> 5;   // 0..3
  const int lane = threadIdx.x & 31;

  for (int t = 0; t < T_; ++t) {
    const _Float16* hv_in = (t & 1) ? hv1 : hv0;
    _Float16*       hv_nw = (t & 1) ? hv0 : hv1;
    const _Float16* hm_in = (t & 1) ? hm1 : hm0;
    _Float16*       hm_nw = (t & 1) ? hm0 : hm1;
    const float*    xt    = data + t * I_;   // row stride T_*I_

    // ---------- Phase A: hv_nw = retanh(x_t@Wi2h^T + hv_in@Wh2h^T + bi2h + bh2h)
    {
      int m0 = wave * 16, n0 = g * 16;
      v8f acc = {};
#pragma unroll 4
      for (int k = 0; k < I_; k += 32)
        acc = wmma_f16(acc, load_a_f32(xt, T_ * I_, m0, k, lane),
                            load_b_f16(Wi2h, I_, n0, k, lane));
#pragma unroll 4
      for (int k = 0; k < H_; k += 32)
        acc = wmma_f16(acc, load_a_f16(hv_in, H_, m0, k, lane),
                            load_b_f16(Wh2h, H_, n0, k, lane));
      int n  = n0 + (lane & 15);
      int mb = m0 + ((lane & 16) ? 8 : 0);
      float bias = bi2h[n] + bh2h[n];
#pragma unroll
      for (int v = 0; v < 8; ++v)
        hv_nw[(mb + v) * H_ + n] = (_Float16)retanh(acc[v] + bias);
    }
    gsync(bar);

    // ---------- Phase B1: out_v = hv_nw@Wh2o^T + bh2o   (split-K over 4 waves)
    {
      int m0 = (g & 3) * 16, n0 = (g >> 2) * 16;
      v8f acc = {};
      int kb = wave * (H_ / 4);
#pragma unroll 4
      for (int k = kb; k < kb + H_ / 4; k += 32)
        acc = wmma_f16(acc, load_a_f16(hv_nw, H_, m0, k, lane),
                            load_b_f16(Wh2o, H_, n0, k, lane));
      reduce4_store_f16(acc, red, wave, lane, out_v, O_, m0, n0, bh2o, false);
    }
    gsync(bar);

    // ---------- Phase B2: out_t = retanh(out_v@Wthal^T + bthal)
    {
      int m0 = (g & 3) * 16, n0 = (g >> 2) * 16;
      v8f acc = {};
      int kb = wave * (O_ / 4);
#pragma unroll
      for (int k = kb; k < kb + O_ / 4; k += 32)
        acc = wmma_f16(acc, load_a_f16(out_v, O_, m0, k, lane),
                            load_b_f16(Wthal, O_, n0, k, lane));
      reduce4_store_f16(acc, red, wave, lane, out_t, O_, m0, n0, bthal, true);
    }
    gsync(bar);

    // ---------- Phase C: hm_nw = retanh(out_t@Wi2h_dos^T + hm_in@Wh2h_dos^T + b)
    {
      int m0 = wave * 16, n0 = g * 16;
      v8f acc = {};
#pragma unroll 4
      for (int k = 0; k < O_; k += 32)
        acc = wmma_f16(acc, load_a_f16(out_t, O_, m0, k, lane),
                            load_b_f16(Wi2h_dos, O_, n0, k, lane));
#pragma unroll 4
      for (int k = 0; k < H_; k += 32)
        acc = wmma_f16(acc, load_a_f16(hm_in, H_, m0, k, lane),
                            load_b_f16(Wh2h_dos, H_, n0, k, lane));
      int n  = n0 + (lane & 15);
      int mb = m0 + ((lane & 16) ? 8 : 0);
      float bias = bi2h_dos[n] + bh2h_dos[n];
#pragma unroll
      for (int v = 0; v < 8; ++v)
        hm_nw[(mb + v) * H_ + n] = (_Float16)retanh(acc[v] + bias);
    }
    gsync(bar);

    // ---------- Phase D: out_m[:, t, :] = hm_nw@Wh2o_dos^T + bh2o_dos (f32 out)
    {
      int m0 = (g & 3) * 16, n0 = (g >> 2) * 16;
      v8f acc = {};
      int kb = wave * (H_ / 4);
#pragma unroll 4
      for (int k = kb; k < kb + H_ / 4; k += 32)
        acc = wmma_f16(acc, load_a_f16(hm_nw, H_, m0, k, lane),
                            load_b_f16(Wh2o_dos, H_, n0, k, lane));
      reduce4_store_out(acc, red, wave, lane, out, t, m0, n0, bh2o_dos);
    }
    // No barrier needed before next step's Phase A (disjoint buffers; A's own
    // barrier orders hv_nw visibility, and each WG passes D before arriving there).
  }
}

__global__ void cvt_f32_f16(const float* __restrict__ src, _Float16* __restrict__ dst, int n) {
  int i = blockIdx.x * blockDim.x + threadIdx.x;
  if (i < n) dst[i] = (_Float16)src[i];
}

extern "C" void kernel_launch(void* const* d_in, const int* in_sizes, int n_in,
                              void* d_out, int out_size, void* d_ws, size_t ws_size,
                              hipStream_t stream) {
  const float* data     = (const float*)d_in[0];
  const float* h0_v     = (const float*)d_in[1];
  const float* h0_m     = (const float*)d_in[2];
  const float* Wi2h     = (const float*)d_in[3];
  const float* bi2h     = (const float*)d_in[4];
  const float* Wh2h     = (const float*)d_in[5];
  const float* bh2h     = (const float*)d_in[6];
  const float* Wh2o     = (const float*)d_in[7];
  const float* bh2o     = (const float*)d_in[8];
  const float* Wthal    = (const float*)d_in[9];
  const float* bthal    = (const float*)d_in[10];
  const float* Wi2h_dos = (const float*)d_in[11];
  const float* bi2h_dos = (const float*)d_in[12];
  const float* Wh2h_dos = (const float*)d_in[13];
  const float* bh2h_dos = (const float*)d_in[14];
  const float* Wh2o_dos = (const float*)d_in[15];
  const float* bh2o_dos = (const float*)d_in[16];
  (void)in_sizes; (void)n_in; (void)out_size; (void)ws_size;

  // Scratch layout: [0,64): barrier counter; then f16 arrays (16B aligned).
  unsigned* bar   = (unsigned*)d_ws;
  _Float16* hbase = (_Float16*)((char*)d_ws + 64);
  size_t off = 0;
  _Float16* wi2h_h     = hbase + off; off += (size_t)H_ * I_;
  _Float16* wh2h_h     = hbase + off; off += (size_t)H_ * H_;
  _Float16* wh2o_h     = hbase + off; off += (size_t)O_ * H_;
  _Float16* wthal_h    = hbase + off; off += (size_t)O_ * O_;
  _Float16* wi2h_dos_h = hbase + off; off += (size_t)H_ * O_;
  _Float16* wh2h_dos_h = hbase + off; off += (size_t)H_ * H_;
  _Float16* wh2o_dos_h = hbase + off; off += (size_t)O_ * H_;
  _Float16* hv0        = hbase + off; off += (size_t)B_ * H_;
  _Float16* hv1        = hbase + off; off += (size_t)B_ * H_;
  _Float16* hm0        = hbase + off; off += (size_t)B_ * H_;
  _Float16* hm1        = hbase + off; off += (size_t)B_ * H_;
  _Float16* outv       = hbase + off; off += (size_t)B_ * O_;
  _Float16* outt       = hbase + off; off += (size_t)B_ * O_;

  hipMemsetAsync(bar, 0, 64, stream);  // reset barrier every launch (graph-safe)

  auto cvt = [&](const float* s, _Float16* d, int n) {
    cvt_f32_f16<<<(n + 255) / 256, 256, 0, stream>>>(s, d, n);
  };
  cvt(Wi2h,     wi2h_h,     H_ * I_);
  cvt(Wh2h,     wh2h_h,     H_ * H_);
  cvt(Wh2o,     wh2o_h,     O_ * H_);
  cvt(Wthal,    wthal_h,    O_ * O_);
  cvt(Wi2h_dos, wi2h_dos_h, H_ * O_);
  cvt(Wh2h_dos, wh2h_dos_h, H_ * H_);
  cvt(Wh2o_dos, wh2o_dos_h, O_ * H_);
  cvt(h0_v, hv0, B_ * H_);
  cvt(h0_m, hm0, B_ * H_);

  millies_rnn_persistent<<<NWG, NTHR, 0, stream>>>(
      data, bi2h, bh2h, bh2o, bthal, bi2h_dos, bh2h_dos, bh2o_dos,
      wi2h_h, wh2h_h, wh2o_h, wthal_h, wi2h_dos_h, wh2h_dos_h, wh2o_dos_h,
      hv0, hv1, hm0, hm1, outv, outt,
      (float*)d_out, bar);
}
